// TanhLSTMCell_68521908240998
// MI455X (gfx1250) — matrix operand order repaired and checked by reading, not compiled, not run on hardware
//
#include <hip/hip_runtime.h>
#include <math.h>

// Problem constants (match reference)
#define B_SZ   16384
#define W_SZ   8
#define IN_SZ  8
#define NF_SZ  5
#define NQ_SZ  64
#define H_SZ   256
#define HD_SZ  256
#define FFT_DIM 80
#define FEAT_PAD 96      // 80 padded to multiple of 32
#define PI_F 3.14159265358979f

typedef __attribute__((ext_vector_type(16))) _Float16 v16h;
typedef __attribute__((ext_vector_type(8)))  _Float16 v8h;
typedef __attribute__((ext_vector_type(8)))  float    v8f;

// ---------------------------------------------------------------------------
// f32 -> f16 convert with 2-D zero padding (used for weights).
// ---------------------------------------------------------------------------
__global__ void cvt_pad_kernel(const float* __restrict__ src, _Float16* __restrict__ dst,
                               int Rs, int Cs, int Rd, int Cd) {
    int idx = blockIdx.x * blockDim.x + threadIdx.x;
    int total = Rd * Cd;
    if (idx >= total) return;
    int r = idx / Cd;
    int c = idx - r * Cd;
    float v = (r < Rs && c < Cs) ? src[(size_t)r * Cs + c] : 0.0f;
    dst[idx] = (_Float16)v;
}

// ---------------------------------------------------------------------------
// 8-point real DFT features: mag=log1p(|X_k|), phase=angle(X_k)/pi, k=0..4.
// feats layout per row: [mag(5x8), phase(5x8), zero pad to 96], stored f16.
// ---------------------------------------------------------------------------
__global__ void feat_kernel(const float* __restrict__ x, _Float16* __restrict__ feats) {
    int b = blockIdx.x * blockDim.x + threadIdx.x;
    if (b >= B_SZ) return;
    float xv[64];
#pragma unroll
    for (int i = 0; i < 64; ++i) xv[i] = x[(size_t)b * 64 + i];

    const float ct[8] = {1.f, 0.70710678f, 0.f, -0.70710678f, -1.f, -0.70710678f, 0.f, 0.70710678f};
    const float st[8] = {0.f, 0.70710678f, 1.f, 0.70710678f, 0.f, -0.70710678f, -1.f, -0.70710678f};

    _Float16* out = feats + (size_t)b * FEAT_PAD;
#pragma unroll
    for (int k = 0; k < NF_SZ; ++k) {
#pragma unroll
        for (int i = 0; i < IN_SZ; ++i) {
            float re = 0.f, im = 0.f;
#pragma unroll
            for (int n = 0; n < W_SZ; ++n) {
                int m = (k * n) & 7;
                float v = xv[n * 8 + i];
                re += v * ct[m];
                im -= v * st[m];
            }
            float mag = log1pf(sqrtf(re * re + im * im));
            float ph  = atan2f(im, re) * (1.0f / PI_F);
            out[k * 8 + i]      = (_Float16)mag;
            out[40 + k * 8 + i] = (_Float16)ph;
        }
    }
#pragma unroll
    for (int j = FFT_DIM; j < FEAT_PAD; ++j) out[j] = (_Float16)0.f;
}

// ---------------------------------------------------------------------------
// h_in = concat(c_mag, c_phase) -> f16 [B, 512]
// ---------------------------------------------------------------------------
__global__ void hin_kernel(const float* __restrict__ c_mag, const float* __restrict__ c_phase,
                           _Float16* __restrict__ hin) {
    int idx = blockIdx.x * blockDim.x + threadIdx.x;
    if (idx >= B_SZ * 2 * H_SZ) return;
    int b = idx >> 9;          // /512
    int j = idx & 511;
    float v = (j < H_SZ) ? c_mag[(size_t)b * H_SZ + j] : c_phase[(size_t)b * H_SZ + j - H_SZ];
    hin[idx] = (_Float16)v;
}

// ---------------------------------------------------------------------------
// Register-blocked WMMA GEMM: out = epilogue(A[M,K] * W[N,K]^T + bias[N])
//   - wave32; each wave computes an MT x NT grid of 16x16 D tiles via
//     v_wmma_f32_16x16x32_f16, reusing A fragments across NT tiles and
//     B fragments across MT tiles (per 32-K chunk: 2*(MT+NT) b128 loads
//     for MT*NT WMMAs).
//   - blockDim = 128 (4 waves stacked on M), gridDim.y = N/(16*NT),
//     gridDim.z = gate batch (per-gate strides on A/W/bias/out)
//   - epilogue: + bias, optional tanh*scale, optional f32 addend,
//     stores f32 and/or f16, columns guarded by Nactual (for N=8 output pad)
// A fragment lane layout (ISA 7.12.2, 16-bit A 16x32):
//   lanes 0-15:  row=m0+lane, K = kk+{0..7} and kk+{16..23}
//   lanes 16-31: row=m0+lane-16, K = kk+{8..15} and kk+{24..31}
// B fragment (W is [N,K] row-major, B = W^T):
//   lanes 0-15:  col n0+lane, K = kk+{0..7} and kk+{16..23}; lanes 16-31: K = kk+{8..15} and kk+{24..31}
// ---------------------------------------------------------------------------
template <int MT, int NT>
__global__ __launch_bounds__(128) void wmma_gemm_kernel(
    const _Float16* __restrict__ A, int Astride,
    const _Float16* __restrict__ Wt, int Wstride,
    const float* __restrict__ bias, int biasStride,
    const float* __restrict__ addend, int ldadd,
    float* __restrict__ outF32, int ldcF32, int o32Stride,
    _Float16* __restrict__ outF16, int ldcF16, int o16Stride,
    int lda, int K, int Nactual, int applyTanh, float scale)
{
    const int g = blockIdx.z;
    A    += (size_t)g * Astride;
    Wt   += (size_t)g * Wstride;
    bias += (size_t)g * biasStride;

    const int lane   = threadIdx.x & 31;
    const int wave   = threadIdx.x >> 5;
    const int laneLo = lane & 15;
    const int laneHi = lane >> 4;

    const int m0 = (blockIdx.x * 4 + wave) * (16 * MT);
    const int n0 = blockIdx.y * (16 * NT);

    const _Float16* aRow[MT];
    const _Float16* wRow[NT];
#pragma unroll
    for (int mt = 0; mt < MT; ++mt)
        aRow[mt] = A + (size_t)(m0 + mt * 16 + laneLo) * lda + laneHi * 8;
#pragma unroll
    for (int nt = 0; nt < NT; ++nt)
        wRow[nt] = Wt + (size_t)(n0 + nt * 16 + laneLo) * K + laneHi * 8;

    v8f acc[MT][NT] = {};
    for (int kk = 0; kk < K; kk += 32) {
        v16h a[MT];
        v16h b[NT];
#pragma unroll
        for (int mt = 0; mt < MT; ++mt) {
            v8h a0 = *(const v8h*)(aRow[mt] + kk);       // 128-bit load
            v8h a1 = *(const v8h*)(aRow[mt] + kk + 16);  // 128-bit load
#pragma unroll
            for (int i = 0; i < 8; ++i) { a[mt][i] = a0[i]; a[mt][i + 8] = a1[i]; }
        }
#pragma unroll
        for (int nt = 0; nt < NT; ++nt) {
            v8h b0 = *(const v8h*)(wRow[nt] + kk);
            v8h b1 = *(const v8h*)(wRow[nt] + kk + 16);
#pragma unroll
            for (int i = 0; i < 8; ++i) { b[nt][i] = b0[i]; b[nt][i + 8] = b1[i]; }
        }
#pragma unroll
        for (int mt = 0; mt < MT; ++mt)
#pragma unroll
            for (int nt = 0; nt < NT; ++nt)
                acc[mt][nt] = __builtin_amdgcn_wmma_f32_16x16x32_f16(
                    /*neg_a=*/false, a[mt], /*neg_b=*/false, b[nt],
                    /*c_mod=*/(short)0, acc[mt][nt],
                    /*reuse_a=*/false, /*reuse_b=*/false);
    }

#pragma unroll
    for (int nt = 0; nt < NT; ++nt) {
        const int  col   = n0 + nt * 16 + laneLo;
        const bool colOk = (col < Nactual);
        const float bv   = colOk ? bias[col] : 0.0f;
#pragma unroll
        for (int mt = 0; mt < MT; ++mt) {
#pragma unroll
            for (int r = 0; r < 8; ++r) {
                const int row = m0 + mt * 16 + r + laneHi * 8;  // C layout: vgpr r -> M=r / M=8+r
                float v = acc[mt][nt][r] + bv;
                if (applyTanh) v = tanhf(v) * scale;
                if (colOk) {
                    if (addend) v += addend[(size_t)row * ldadd + col];
                    if (outF32) outF32[(size_t)g * o32Stride + (size_t)row * ldcF32 + col] = v;
                    if (outF16) outF16[(size_t)g * o16Stride + (size_t)row * ldcF16 + col] = (_Float16)v;
                }
            }
        }
    }
}

// ---------------------------------------------------------------------------
// Gate elementwise: consumes g[4,B,H] (tanh'd, f16), c_mag, c_phase;
// produces c_mag_new, c_phase_new (f32, straight into d_out) and h_t (f16).
// ---------------------------------------------------------------------------
__global__ void gate_kernel(const _Float16* __restrict__ gbuf,
                            const float* __restrict__ c_mag,
                            const float* __restrict__ c_phase,
                            float* __restrict__ c_mag_out,
                            float* __restrict__ c_phase_out,
                            _Float16* __restrict__ ht) {
    int idx = blockIdx.x * blockDim.x + threadIdx.x;
    const int BH = B_SZ * H_SZ;
    if (idx >= BH) return;
    float gi = (float)gbuf[idx];
    float gf = (float)gbuf[BH + idx];
    float gg = (float)gbuf[2 * BH + idx];
    float go = (float)gbuf[3 * BH + idx];
    float i_t = (gi + 1.f) * 0.5f;
    float f_t = (gf + 1.f) * 0.5f;
    float o_t = (go + 1.f) * 0.5f;
    float cm = c_mag[idx];
    float cp = c_phase[idx];
    float cmn = f_t * cm + i_t * fabsf(gg);
    float ang = gg * PI_F;
    float wrapped = atan2f(sinf(ang), cosf(ang)) * (1.0f / PI_F);
    float cpn = cp + i_t * wrapped;
    cpn = cpn + 1.0f;
    cpn = cpn - 2.0f * floorf(cpn * 0.5f);   // remainder(x, 2) in [0,2)
    cpn -= 1.0f;
    float h = o_t * cmn * cosf(cpn * PI_F);
    c_mag_out[idx]   = cmn;
    c_phase_out[idx] = cpn;
    ht[idx] = (_Float16)h;
}

// ---------------------------------------------------------------------------
// Host-side templated GEMM launcher
// ---------------------------------------------------------------------------
template <int MT, int NT>
static void launch_gemm(hipStream_t stream,
                        const _Float16* A, int Astride, int lda,
                        const _Float16* Wt, int Wstride, int K,
                        const float* bias, int biasStride,
                        const float* addend, int ldadd,
                        float* o32, int ldc32, int o32s,
                        _Float16* o16, int ldc16, int o16s,
                        int N, int Nactual, int G, int doTanh, float scale) {
    dim3 grid(B_SZ / (64 * MT), N / (16 * NT), G);
    wmma_gemm_kernel<MT, NT><<<grid, dim3(128), 0, stream>>>(
        A, Astride, Wt, Wstride, bias, biasStride, addend, ldadd,
        o32, ldc32, o32s, o16, ldc16, o16s, lda, K, Nactual, doTanh, scale);
}

// ---------------------------------------------------------------------------
// Host-side launch
// ---------------------------------------------------------------------------
extern "C" void kernel_launch(void* const* d_in, const int* in_sizes, int n_in,
                              void* d_out, int out_size, void* d_ws, size_t ws_size,
                              hipStream_t stream) {
    (void)in_sizes; (void)n_in; (void)out_size; (void)ws_size;

    const float* x       = (const float*)d_in[0];
    const float* c_mag   = (const float*)d_in[1];
    const float* c_phase = (const float*)d_in[2];
    const float* Wip     = (const float*)d_in[3];
    const float* bip     = (const float*)d_in[4];
    const float* Whp     = (const float*)d_in[5];
    const float* bhp     = (const float*)d_in[6];
    const float* Wg1     = (const float*)d_in[7];
    const float* bg1     = (const float*)d_in[8];
    const float* Wg2     = (const float*)d_in[9];
    const float* bg2     = (const float*)d_in[10];
    const float* Wgo     = (const float*)d_in[11];
    const float* bgo     = (const float*)d_in[12];
    const float* Wout    = (const float*)d_in[13];
    const float* bout    = (const float*)d_in[14];

    // Output layout: y [B,8] | c_mag_new [B,256] | c_phase_new [B,256]
    float* y_out  = (float*)d_out;
    float* cm_out = y_out + (size_t)B_SZ * IN_SZ;
    float* cp_out = cm_out + (size_t)B_SZ * H_SZ;

    // Bump allocator over d_ws (deterministic, no hipMalloc)
    char* ws = (char*)d_ws;
    size_t off = 0;
    auto alloc = [&](size_t bytes) -> void* {
        void* p = ws + off;
        off = (off + bytes + 255) & ~(size_t)255;
        return p;
    };
    const int BH = B_SZ * H_SZ;
    _Float16* featsA   = (_Float16*)alloc((size_t)B_SZ * FEAT_PAD * 2);
    _Float16* hinA     = (_Float16*)alloc((size_t)B_SZ * 2 * H_SZ * 2);   // reused as h_t
    float*    proj     = (float*)   alloc((size_t)B_SZ * NQ_SZ * 4);
    _Float16* combined = (_Float16*)alloc((size_t)B_SZ * NQ_SZ * 2);
    _Float16* h1       = (_Float16*)alloc((size_t)4 * BH * 2);            // reused as gbuf
    _Float16* h2       = (_Float16*)alloc((size_t)4 * BH * 2);
    _Float16* Wiph     = (_Float16*)alloc((size_t)NQ_SZ * FEAT_PAD * 2);
    _Float16* Whph     = (_Float16*)alloc((size_t)NQ_SZ * 2 * H_SZ * 2);
    _Float16* Wg1h     = (_Float16*)alloc((size_t)4 * HD_SZ * NQ_SZ * 2);
    _Float16* Wg2h     = (_Float16*)alloc((size_t)4 * HD_SZ * HD_SZ * 2);
    _Float16* Wgoh     = (_Float16*)alloc((size_t)4 * H_SZ * HD_SZ * 2);
    _Float16* Wouth    = (_Float16*)alloc((size_t)16 * H_SZ * 2);
    _Float16* gbuf = h1;
    _Float16* ht   = hinA;

    auto cvt = [&](const float* src, _Float16* dst, int Rs, int Cs, int Rd, int Cd) {
        int n = Rd * Cd;
        cvt_pad_kernel<<<dim3((n + 255) / 256), dim3(256), 0, stream>>>(src, dst, Rs, Cs, Rd, Cd);
    };
    // Weight conversions (+padding where K/N is not WMMA-friendly)
    cvt(Wip,  Wiph,  NQ_SZ, FFT_DIM, NQ_SZ, FEAT_PAD);          // pad K 80->96
    cvt(Whp,  Whph,  NQ_SZ, 2 * H_SZ, NQ_SZ, 2 * H_SZ);
    cvt(Wg1,  Wg1h,  4 * HD_SZ, NQ_SZ, 4 * HD_SZ, NQ_SZ);
    cvt(Wg2,  Wg2h,  4 * HD_SZ, HD_SZ, 4 * HD_SZ, HD_SZ);
    cvt(Wgo,  Wgoh,  4 * H_SZ,  HD_SZ, 4 * H_SZ,  HD_SZ);
    cvt(Wout, Wouth, IN_SZ, H_SZ, 16, H_SZ);                    // pad N 8->16 (zero rows)

    // FFT features + h_in concat
    feat_kernel<<<dim3((B_SZ + 255) / 256), dim3(256), 0, stream>>>(x, featsA);
    hin_kernel<<<dim3((B_SZ * 2 * H_SZ + 255) / 256), dim3(256), 0, stream>>>(c_mag, c_phase, hinA);

    // x_proj = tanh(feats @ Wip^T + bip) * pi            -> f32 [B,64]
    launch_gemm<2, 4>(stream, featsA, 0, FEAT_PAD, Wiph, 0, FEAT_PAD, bip, 0, nullptr, 0,
                      proj, NQ_SZ, 0, nullptr, 0, 0, NQ_SZ, NQ_SZ, 1, 1, PI_F);
    // combined = tanh(h_in @ Whp^T + bhp) * pi + x_proj  -> f16 [B,64]
    launch_gemm<2, 4>(stream, hinA, 0, 2 * H_SZ, Whph, 0, 2 * H_SZ, bhp, 0, proj, NQ_SZ,
                      nullptr, 0, 0, combined, NQ_SZ, 0, NQ_SZ, NQ_SZ, 1, 1, PI_F);
    // h1 = tanh(combined @ Wg1[g]^T + bg1[g])            -> f16 [4,B,256]
    launch_gemm<2, 4>(stream, combined, 0, NQ_SZ, Wg1h, HD_SZ * NQ_SZ, NQ_SZ, bg1, HD_SZ,
                      nullptr, 0, nullptr, 0, 0, h1, HD_SZ, BH, HD_SZ, HD_SZ, 4, 1, 1.0f);
    // h2 = tanh(h1 @ Wg2[g]^T + bg2[g])                  -> f16 [4,B,256]
    launch_gemm<2, 4>(stream, h1, BH, HD_SZ, Wg2h, HD_SZ * HD_SZ, HD_SZ, bg2, HD_SZ,
                      nullptr, 0, nullptr, 0, 0, h2, HD_SZ, BH, HD_SZ, HD_SZ, 4, 1, 1.0f);
    // g  = tanh(h2 @ Wgo[g]^T + bgo[g])                  -> f16 [4,B,256] (reuses h1)
    launch_gemm<2, 4>(stream, h2, BH, HD_SZ, Wgoh, H_SZ * HD_SZ, HD_SZ, bgo, H_SZ,
                      nullptr, 0, nullptr, 0, 0, gbuf, H_SZ, BH, H_SZ, H_SZ, 4, 1, 1.0f);

    // Gate math -> c_mag_new, c_phase_new (d_out), h_t (f16, reuses hinA)
    gate_kernel<<<dim3((BH + 255) / 256), dim3(256), 0, stream>>>(
        gbuf, c_mag, c_phase, cm_out, cp_out, ht);

    // y = h_t @ Wout^T + bout                            -> f32 [B,8] (N padded to 16)
    launch_gemm<4, 1>(stream, ht, 0, H_SZ, Wouth, 0, H_SZ, bout, 0, nullptr, 0,
                      y_out, IN_SZ, 0, nullptr, 0, 0, 16, IN_SZ, 1, 0, 1.0f);
}
